// GATWithPool_50749333570052
// MI455X (gfx1250) — matrix operand order repaired and checked by reading, not compiled, not run on hardware
//
#include <hip/hip_runtime.h>
#include <hip/hip_bf16.h>

// ---------------- problem constants (from reference) ----------------
#define NN_N   2048
#define D_IN   128
#define D_HID  128
#define HEADS  4
#define D_OUT  10
#define NGRP   8
#define NEG_SLOPE 0.2f

typedef __attribute__((ext_vector_type(16))) __bf16 v16bf;
typedef __attribute__((ext_vector_type(8)))  __bf16 v8bf;
typedef __attribute__((ext_vector_type(4)))  __bf16 v4bf;
typedef __attribute__((ext_vector_type(8)))  float  v8f;

// two contiguous 16-byte LDS chunks -> one 16-element bf16 fragment
static __device__ inline v16bf load_frag(const __bf16* p0, const __bf16* p1) {
    v8bf a = *(const v8bf*)p0;
    v8bf b = *(const v8bf*)p1;
    return __builtin_shufflevector(a, b, 0,1,2,3,4,5,6,7,8,9,10,11,12,13,14,15);
}

// =====================================================================
// 1) agg[s,t] = sum_c conv_w[c] * attn[c,s,t] + conv_b   (201 MB pass)
// =====================================================================
__global__ void k_agg(const float* __restrict__ attn, const float* __restrict__ w,
                      const float* __restrict__ b, float* __restrict__ agg) {
    const long long NN = (long long)NN_N * NN_N;
    long long i = (long long)blockIdx.x * blockDim.x + threadIdx.x;
    if (i >= NN) return;
    float a = b[0];
#pragma unroll
    for (int c = 0; c < 12; ++c) a += w[c] * attn[c * NN + i];
    agg[i] = a;
}

// =====================================================================
// 2) per-target mean of incoming off-diagonal positive edges
// =====================================================================
__global__ void k_colstats(const float* __restrict__ agg, float* __restrict__ mean_attr) {
    int t = blockIdx.x * blockDim.x + threadIdx.x;
    if (t >= NN_N) return;
    float sum = 0.f; int cnt = 0;
    for (int s = 0; s < NN_N; ++s) {
        __builtin_prefetch(&agg[(s + 16) * NN_N + t], 0, 0);   // global_prefetch_b8
        float av = agg[s * NN_N + t];                          // coalesced across t
        if (av > 0.f && s != t) { sum += av; ++cnt; }
    }
    mean_attr[t] = sum / (float)(cnt > 1 ? cnt : 1);
}

// =====================================================================
// 3) c_e[h] = dot(a_edge[h,:], We[h,:])
// =====================================================================
__global__ void k_ce(const float* __restrict__ a_edge, const float* __restrict__ We,
                     float* __restrict__ ce) {
    __shared__ float red[128];
    int h = blockIdx.x, tid = threadIdx.x;
    red[tid] = a_edge[h * D_HID + tid] * We[h * D_HID + tid];
    __syncthreads();
    for (int st = 64; st > 0; st >>= 1) { if (tid < st) red[tid] += red[tid + st]; __syncthreads(); }
    if (tid == 0) ce[h] = red[0];
}

// =====================================================================
// 4) tiled bf16 WMMA GEMM: C[M,Nc] = A[M,K] * B[K,Nc]  (fp32 in/out)
//    block = 128 threads (4 waves), tile 64x64, K-step 32
//    A tile LDS: [m][k] row-major; B tile LDS: [n][k] TRANSPOSED so that
//    per-lane fragments are contiguous -> ds_load_b128 pairs.
// =====================================================================
__global__ void k_gemm_bf16(const float* __restrict__ A, const float* __restrict__ B,
                            float* __restrict__ C, int M, int K, int Nc) {
    __shared__ __bf16 As[64][32];    // [m][k]
    __shared__ __bf16 Bs[64][32];    // [n][k]  (transposed)
    const int tid  = threadIdx.x;
    const int wave = tid >> 5;
    const int lane = tid & 31;
    const int l15  = lane & 15;
    const int hi   = lane >> 4;
    const int m0   = blockIdx.x * 64;
    const int n0   = blockIdx.y * 64;

    v8f acc[4];
#pragma unroll
    for (int j = 0; j < 4; ++j)
#pragma unroll
        for (int r = 0; r < 8; ++r) acc[j][r] = 0.f;

    const int arow  = wave * 16 + l15;
    const int kbase = hi * 8;        // A frag: K {kbase..kbase+7, kbase+16..kbase+23}
    const int kb16  = hi * 16;       // B frag: same K as A frag

    for (int k0 = 0; k0 < K; k0 += 32) {
        // stage A 64x32 (float4 global loads, v4bf LDS stores)
#pragma unroll
        for (int e = 0; e < 4; ++e) {
            int flat = (e * 128 + tid) * 4;
            int r = flat >> 5, c = flat & 31;
            float4 v = *(const float4*)&A[(m0 + r) * K + k0 + c];
            v4bf w = { (__bf16)v.x, (__bf16)v.y, (__bf16)v.z, (__bf16)v.w };
            *(v4bf*)&As[r][c] = w;
        }
        // stage B 32x64 -> transposed [n][k]
#pragma unroll
        for (int e = 0; e < 4; ++e) {
            int flat = (e * 128 + tid) * 4;
            int r = flat >> 6, c = flat & 63;            // r = k, c = n (coalesced)
            float4 v = *(const float4*)&B[(k0 + r) * Nc + n0 + c];
            Bs[c + 0][r] = (__bf16)v.x;
            Bs[c + 1][r] = (__bf16)v.y;
            Bs[c + 2][r] = (__bf16)v.z;
            Bs[c + 3][r] = (__bf16)v.w;
        }
        __syncthreads();

        v16bf afrag = load_frag(&As[arow][kbase], &As[arow][kbase + 16]);
#pragma unroll
        for (int j = 0; j < 4; ++j) {
            const int col = j * 16 + l15;
            v16bf bfrag = load_frag(&Bs[col][kbase], &Bs[col][kbase + 16]);
            acc[j] = __builtin_amdgcn_wmma_f32_16x16x32_bf16(
                false, afrag, false, bfrag, (short)0, acc[j], false, false);
        }
        __syncthreads();
    }
#pragma unroll
    for (int j = 0; j < 4; ++j) {
        int col = n0 + j * 16 + l15;
#pragma unroll
        for (int r = 0; r < 8; ++r)
            C[(m0 + wave * 16 + r + hi * 8) * Nc + col] = acc[j][r];
    }
}

// =====================================================================
// 5) per-row attention scores: s_src[n,h], s_dst[n,h]
// =====================================================================
__global__ void k_rowdot(const float* __restrict__ hmat, const float* __restrict__ a_src,
                         const float* __restrict__ a_dst, float* __restrict__ s_src,
                         float* __restrict__ s_dst, int Hh) {
    int wv   = (blockIdx.x * blockDim.x + threadIdx.x) >> 5;
    int lane = threadIdx.x & 31;
    if (wv >= NN_N) return;
    for (int h = 0; h < Hh; ++h) {
        float vs = 0.f, vd = 0.f;
        for (int c = lane; c < D_HID; c += 32) {
            float hv = hmat[wv * (Hh * D_HID) + h * D_HID + c];
            vs += hv * a_src[h * D_HID + c];
            vd += hv * a_dst[h * D_HID + c];
        }
        for (int off = 16; off > 0; off >>= 1) {
            vs += __shfl_down(vs, off, 32);
            vd += __shfl_down(vd, off, 32);
        }
        if (lane == 0) { s_src[wv * Hh + h] = vs; s_dst[wv * Hh + h] = vd; }
    }
}

// =====================================================================
// 6) fp32 [n][h*C+c] -> bf16 head-major [h][n][c]
// =====================================================================
__global__ void k_convert_headmajor(const float* __restrict__ hmat, __bf16* __restrict__ hbf, int Hh) {
    int i = blockIdx.x * blockDim.x + threadIdx.x;
    if (i >= NN_N * Hh * D_HID) return;
    int c = i % D_HID; int rest = i / D_HID; int h = rest % Hh; int n = rest / Hh;
    hbf[(h * NN_N + n) * D_HID + c] = (__bf16)hmat[i];
}

// =====================================================================
// 7) softmax pass 1: per (target,head) online max & sumexp over sources
// =====================================================================
__global__ void k_attn_stats(const float* __restrict__ agg, const float* __restrict__ mean_attr,
                             const float* __restrict__ s_src, const float* __restrict__ s_dst,
                             const float* __restrict__ ce, float* __restrict__ mstat,
                             float* __restrict__ lstat, int Hh) {
    int t = blockIdx.x * blockDim.x + threadIdx.x;
    if (t >= NN_N) return;
    const float ma = mean_attr[t];
    float sd[HEADS], m[HEADS], l[HEADS], c[HEADS];
    for (int h = 0; h < Hh; ++h) { sd[h] = s_dst[t*Hh+h]; m[h] = -1e30f; l[h] = 0.f; c[h] = ce[h]; }
    for (int s = 0; s < NN_N; ++s) {
        __builtin_prefetch(&agg[(s + 16) * NN_N + t], 0, 0);
        float av   = agg[s * NN_N + t];
        bool moff  = (av > 0.f) && (s != t);
        bool adj   = moff || (s == t);
        if (!adj) continue;
        float ea = (moff ? av : 0.f) + ((s == t) ? ma : 0.f);
        for (int h = 0; h < Hh; ++h) {
            float lg = s_src[s * Hh + h] + sd[h] + ea * c[h];
            lg = lg >= 0.f ? lg : NEG_SLOPE * lg;
            if (lg > m[h]) { l[h] = l[h] * __expf(m[h] - lg) + 1.f; m[h] = lg; }
            else           { l[h] += __expf(lg - m[h]); }
        }
    }
    for (int h = 0; h < Hh; ++h) { mstat[t*Hh+h] = m[h]; lstat[t*Hh+h] = l[h]; }
}

// =====================================================================
// 8) softmax pass 2 + aggregation: out[t,h,:] = (P_unnorm @ H_h)/l + bias
//    block = 256 thr (8 waves), each wave owns a 16-target tile; WMMA over
//    all 2048 sources in K=32 chunks, 8 c-tiles (c=128) accumulated in f32.
//    Hs chunk staged TRANSPOSED [c][s] for contiguous fragments.
// =====================================================================
__global__ void k_attn_agg(const float* __restrict__ agg, const float* __restrict__ mean_attr,
                           const float* __restrict__ s_src, const float* __restrict__ s_dst,
                           const float* __restrict__ ce, const float* __restrict__ mstat,
                           const float* __restrict__ lstat, const __bf16* __restrict__ hbf,
                           const float* __restrict__ bias, float* __restrict__ out,
                           int Hh, int relu) {
    __shared__ __bf16 Hs[D_HID][32];             // [c][s_local], 8 KB
    const int tid  = threadIdx.x;
    const int wave = tid >> 5;
    const int lane = tid & 31;
    const int l15  = lane & 15;
    const int hi   = lane >> 4;
    const int h    = blockIdx.y;
    const int tb   = (blockIdx.x * 8 + wave) * 16;

    const float ceh    = ce[h];
    const int   t_a    = tb + l15;               // A-fragment row's target
    const float sdst_a = s_dst[t_a * Hh + h];
    const float m_a    = mstat[t_a * Hh + h];
    const float mean_a = mean_attr[t_a];

    v8f acc[8];
#pragma unroll
    for (int j = 0; j < 8; ++j)
#pragma unroll
        for (int r = 0; r < 8; ++r) acc[j][r] = 0.f;

    const int kbase = hi * 8;
    const int kb16  = hi * 16;

    for (int s0 = 0; s0 < NN_N; s0 += 32) {
        // cooperative stage of H chunk (v8bf global loads), transposed into LDS
#pragma unroll
        for (int e = 0; e < 2; ++e) {
            int flat = (e * 256 + tid) * 8;
            int r = flat >> 7, c = flat & 127;   // r = s_local, c = channel (coalesced)
            v8bf v = *(const v8bf*)&hbf[(h * NN_N + s0 + r) * D_HID + c];
#pragma unroll
            for (int j = 0; j < 8; ++j) Hs[c + j][r] = v[j];
        }
        __syncthreads();

        // build P fragment (16 targets x 32 sources) directly in registers
        v16bf afrag;
#pragma unroll
        for (int i = 0; i < 16; ++i) {
            int s     = s0 + kbase + (i & 7) + ((i >> 3) << 4);
            float av  = agg[s * NN_N + t_a];
            bool moff = (av > 0.f) && (s != t_a);
            bool adj  = moff || (s == t_a);
            float ea  = (moff ? av : 0.f) + ((s == t_a) ? mean_a : 0.f);
            float lg  = s_src[s * Hh + h] + sdst_a + ea * ceh;
            lg = lg >= 0.f ? lg : NEG_SLOPE * lg;
            float p = adj ? __expf(lg - m_a) : 0.f;
            afrag[i] = (__bf16)p;
        }
#pragma unroll
        for (int ct = 0; ct < 8; ++ct) {
            const int col = ct * 16 + l15;
            v16bf bfrag = load_frag(&Hs[col][kbase], &Hs[col][kbase + 16]);
            acc[ct] = __builtin_amdgcn_wmma_f32_16x16x32_bf16(
                false, afrag, false, bfrag, (short)0, acc[ct], false, false);
        }
        __syncthreads();
    }
#pragma unroll
    for (int r = 0; r < 8; ++r) {
        const int   t     = tb + r + hi * 8;
        const float inv_l = 1.0f / lstat[t * Hh + h];
#pragma unroll
        for (int ct = 0; ct < 8; ++ct) {
            int   c = ct * 16 + l15;
            float v = acc[ct][r] * inv_l + bias[h * D_HID + c];
            if (relu) v = fmaxf(v, 0.f);
            out[t * (Hh * D_HID) + h * D_HID + c] = v;
        }
    }
}

// =====================================================================
// 9) segment mean pool + FC + log-softmax  (one block per group)
// =====================================================================
__global__ void k_pool_fc(const float* __restrict__ x2, const int* __restrict__ bidx,
                          const float* __restrict__ fc_w, const float* __restrict__ fc_b,
                          float* __restrict__ out) {
    __shared__ float pooled[D_HID];
    __shared__ float logits[D_OUT];
    __shared__ float lse;
    const int g = blockIdx.x, tid = threadIdx.x;   // 128 threads
    float sum = 0.f; int cnt = 0;
    for (int n = 0; n < NN_N; ++n)
        if (bidx[n] == g) { sum += x2[n * D_HID + tid]; ++cnt; }
    pooled[tid] = sum / (float)(cnt > 1 ? cnt : 1);
    __syncthreads();
    if (tid < D_OUT) {
        float lg = fc_b[tid];
        for (int c = 0; c < D_HID; ++c) lg += pooled[c] * fc_w[c * D_OUT + tid];
        logits[tid] = lg;
    }
    __syncthreads();
    if (tid == 0) {
        float m = -1e30f;
        for (int o = 0; o < D_OUT; ++o) m = fmaxf(m, logits[o]);
        float s = 0.f;
        for (int o = 0; o < D_OUT; ++o) s += __expf(logits[o] - m);
        lse = m + __logf(s);
    }
    __syncthreads();
    if (tid < D_OUT) out[g * D_OUT + tid] = logits[tid] - lse;
}

// =====================================================================
extern "C" void kernel_launch(void* const* d_in, const int* in_sizes, int n_in,
                              void* d_out, int out_size, void* d_ws, size_t ws_size,
                              hipStream_t stream) {
    const float* x        = (const float*)d_in[0];
    const float* attn     = (const float*)d_in[1];
    const int*   bidx     = (const int*)  d_in[2];
    const float* conv_w   = (const float*)d_in[3];
    const float* conv_b   = (const float*)d_in[4];
    const float* W1       = (const float*)d_in[5];
    const float* att_src1 = (const float*)d_in[6];
    const float* att_dst1 = (const float*)d_in[7];
    const float* att_e1   = (const float*)d_in[8];
    const float* We1      = (const float*)d_in[9];
    const float* b1       = (const float*)d_in[10];
    const float* W2       = (const float*)d_in[11];
    const float* att_src2 = (const float*)d_in[12];
    const float* att_dst2 = (const float*)d_in[13];
    const float* att_e2   = (const float*)d_in[14];
    const float* We2      = (const float*)d_in[15];
    const float* b2       = (const float*)d_in[16];
    const float* fc_w     = (const float*)d_in[17];
    const float* fc_b     = (const float*)d_in[18];
    float* out = (float*)d_out;

    // ---- workspace carve-out (bytes, 256-aligned regions) ----
    char* ws = (char*)d_ws;
    size_t off = 0;
    auto alloc = [&](size_t bytes) { char* p = ws + off; off = (off + bytes + 255) & ~(size_t)255; return p; };
    const size_t NN = (size_t)NN_N * NN_N;
    float*  agg     = (float*) alloc(NN * 4);
    float*  meanat  = (float*) alloc(NN_N * 4);
    float*  ce1     = (float*) alloc(HEADS * 4);
    float*  ce2     = (float*) alloc(4);
    float*  h1      = (float*) alloc((size_t)NN_N * HEADS * D_HID * 4);
    __bf16* h1bf    = (__bf16*)alloc((size_t)NN_N * HEADS * D_HID * 2);
    float*  ssrc1   = (float*) alloc((size_t)NN_N * HEADS * 4);
    float*  sdst1   = (float*) alloc((size_t)NN_N * HEADS * 4);
    float*  m1      = (float*) alloc((size_t)NN_N * HEADS * 4);
    float*  l1      = (float*) alloc((size_t)NN_N * HEADS * 4);
    float*  out1    = (float*) alloc((size_t)NN_N * HEADS * D_HID * 4);
    float*  h2      = (float*) alloc((size_t)NN_N * D_HID * 4);
    __bf16* h2bf    = (__bf16*)alloc((size_t)NN_N * D_HID * 2);
    float*  ssrc2   = (float*) alloc(NN_N * 4);
    float*  sdst2   = (float*) alloc(NN_N * 4);
    float*  m2      = (float*) alloc(NN_N * 4);
    float*  l2      = (float*) alloc(NN_N * 4);
    float*  out2    = (float*) alloc((size_t)NN_N * D_HID * 4);

    // 1) conv over channels (the single 201 MB HBM pass)
    k_agg<<<(int)((NN + 255) / 256), 256, 0, stream>>>(attn, conv_w, conv_b, agg);
    // 2) self-loop mean attrs
    k_colstats<<<NN_N / 256, 256, 0, stream>>>(agg, meanat);
    // 3) edge coefficients
    k_ce<<<HEADS, 128, 0, stream>>>(att_e1, We1, ce1);
    k_ce<<<1,     128, 0, stream>>>(att_e2, We2, ce2);

    // ---- layer 1 ----
    {
        dim3 g(NN_N / 64, (HEADS * D_HID) / 64);
        k_gemm_bf16<<<g, 128, 0, stream>>>(x, W1, h1, NN_N, D_IN, HEADS * D_HID);
    }
    k_rowdot<<<NN_N / 8, 256, 0, stream>>>(h1, att_src1, att_dst1, ssrc1, sdst1, HEADS);
    k_convert_headmajor<<<(NN_N * HEADS * D_HID) / 256, 256, 0, stream>>>(h1, h1bf, HEADS);
    k_attn_stats<<<NN_N / 256, 256, 0, stream>>>(agg, meanat, ssrc1, sdst1, ce1, m1, l1, HEADS);
    {
        dim3 g(NN_N / 128, HEADS);
        k_attn_agg<<<g, 256, 0, stream>>>(agg, meanat, ssrc1, sdst1, ce1, m1, l1,
                                          h1bf, b1, out1, HEADS, /*relu=*/1);
    }

    // ---- layer 2 ----
    {
        dim3 g(NN_N / 64, D_HID / 64);
        k_gemm_bf16<<<g, 128, 0, stream>>>(out1, W2, h2, NN_N, HEADS * D_HID, D_HID);
    }
    k_rowdot<<<NN_N / 8, 256, 0, stream>>>(h2, att_src2, att_dst2, ssrc2, sdst2, 1);
    k_convert_headmajor<<<(NN_N * D_HID) / 256, 256, 0, stream>>>(h2, h2bf, 1);
    k_attn_stats<<<NN_N / 256, 256, 0, stream>>>(agg, meanat, ssrc2, sdst2, ce2, m2, l2, 1);
    {
        dim3 g(NN_N / 128, 1);
        k_attn_agg<<<g, 256, 0, stream>>>(agg, meanat, ssrc2, sdst2, ce2, m2, l2,
                                          h2bf, b2, out2, 1, /*relu=*/0);
    }

    // ---- pool + classify ----
    k_pool_fc<<<NGRP, 128, 0, stream>>>(out2, bidx, fc_w, fc_b, out);
}
